// MultiHeadAttention_14053132992615
// MI455X (gfx1250) — matrix operand order repaired and checked by reading, not compiled, not run on hardware
//
#include <hip/hip_runtime.h>

// ---------------------------------------------------------------------------
// MHA forward for MI455X (gfx1250, wave32, WMMA).
// B=2, T=S=2048, D=1024, H=16, DK=64.
// Pipeline: 3x projection GEMM (f32->bf16, WMMA bf16, f32 acc) -> flash
// attention (WMMA bf16 for QK^T and PV, online softmax) -> output GEMM.
// bf16 staging paths use GLOBAL_LOAD_ASYNC_TO_LDS (ASYNCcnt) when the
// toolchain exposes the builtin; otherwise they fall back to VGPR staging.
// Workspace layout (needs 32 MiB):
//   [0,8M)   Qbf  [B,H,T,64] bf16
//   [8,16M)  Kbf  [B,H,S,64] bf16
//   [16,24M) Vbf  [B,H,S,64] bf16
//   [24,32M) AO   [B,T,D]    bf16  (attention output, heads concatenated)
// ---------------------------------------------------------------------------

typedef __attribute__((ext_vector_type(8)))  __bf16 v8bf;
typedef __attribute__((ext_vector_type(16))) __bf16 v16bf;
typedef __attribute__((ext_vector_type(8)))  float  v8f;

#define DEVINL static __device__ __forceinline__

#if __has_builtin(__builtin_amdgcn_global_load_async_to_lds_b128)
#define HAVE_ASYNC_LDS 1
#else
#define HAVE_ASYNC_LDS 0
#endif

#if HAVE_ASYNC_LDS
// The builtin's pointer parameters are int4-vector typed (per hipcc's
// diagnostic: "__vector_size__(4 * sizeof(int)) int *").
typedef int v4i __attribute__((vector_size(16)));
typedef __attribute__((address_space(1))) v4i gv4i;   // global
typedef __attribute__((address_space(3))) v4i lv4i;   // LDS

// Generic->AS1: value-preserving 64-bit cast. Generic->AS3: the ISA defines
// the LDS address as the low 32 bits of the generic address, so truncating a generic
// pointer into a __shared__ allocation yields the correct 32-bit LDS address.
DEVINL void async_cp_b128(const void* g, void* l) {
  __builtin_amdgcn_global_load_async_to_lds_b128(
      (gv4i*)(unsigned long long)g,
      (lv4i*)(unsigned)(unsigned long long)l,
      0, 0);
}
#endif

DEVINL void wait_async0() {
#if HAVE_ASYNC_LDS
#if __has_builtin(__builtin_amdgcn_s_wait_asynccnt)
  __builtin_amdgcn_s_wait_asynccnt(0);
#else
  asm volatile("s_wait_asynccnt 0" ::: "memory");
#endif
#endif
}

DEVINL __bf16 f2bf(float f) {                 // round-to-nearest-even
  unsigned u = __builtin_bit_cast(unsigned, f);
  unsigned r = (u + 0x7FFFu + ((u >> 16) & 1u)) >> 16;
  unsigned short s = (unsigned short)r;
  return __builtin_bit_cast(__bf16, s);
}

union FragU { v16bf v; v8bf h[2]; };

// A-matrix 16x32 bf16 fragment (ISA 7.12.2): lanes 0-15 hold K in
// [0,8) u [16,24); lanes 16-31 hold K in [8,16) u [24,32).
DEVINL v16bf load_frag_a(const __bf16* p, int hi) {
  FragU f;
  f.h[0] = *(const v8bf*)(p + hi * 8);
  f.h[1] = *(const v8bf*)(p + 16 + hi * 8);
  return f.v;
}
// B-matrix 32x16 bf16 fragment: column-per-lane, lanes 0-15 hold K in
// [0,8) u [16,24); lanes 16-31 hold K in [8,16) u [24,32) (LDS layout [n][k]).
DEVINL v16bf load_frag_b(const __bf16* p, int hi) {
  FragU f;
  f.h[0] = *(const v8bf*)(p + hi * 8);
  f.h[1] = *(const v8bf*)(p + 16 + hi * 8);
  return f.v;
}

DEVINL v8f wmma_bf16(v16bf a, v16bf b, v8f c) {
  return __builtin_amdgcn_wmma_f32_16x16x32_bf16(
      /*neg_a=*/false, a, /*neg_b=*/false, b,
      /*c_mod=*/(short)0, c, /*reuse_a=*/false, /*reuse_b=*/false);
}

DEVINL v8f zero8() { v8f z; for (int i = 0; i < 8; ++i) z[i] = 0.f; return z; }

// ---------------------------------------------------------------------------
// GEMM: C[4096,1024] = A[4096,1024] x W[1024,1024] + bias
// ABF16: 0 = A is fp32 (convert while staging), 1 = A is bf16 in ws
//        (staged via async load-to-LDS when available).
// HEADSPLIT: 1 = write bf16 to [B,H,T,64] (projection), 0 = fp32 [m][n].
// Block tile 128x128, K-tile 64, 8 waves as 2x4, wave tile 64x32 (4x2 WMMAs).
// ---------------------------------------------------------------------------
template <int ABF16, int HEADSPLIT>
__global__ __launch_bounds__(256)
void gemm_wmma(const void* __restrict__ Ap, const float* __restrict__ W,
               const float* __restrict__ bias, void* __restrict__ outp) {
  constexpr int LD = 72;                    // padded k-stride (bank spread)
  constexpr int Kdim = 1024, Ndim = 1024;
  __shared__ __bf16 As[128 * LD];           // [m][k]
  __shared__ __bf16 Bs[128 * LD];           // [n][k]  (transposed tile of W)

  const int tid  = threadIdx.x;
  const int lane = tid & 31, wave = tid >> 5;
  const int l15  = lane & 15, hi = lane >> 4;
  const int wrow = wave >> 2, wcol = wave & 3;        // 2x4 wave grid
  const int Mbase = blockIdx.y * 128;
  const int Nbase = blockIdx.x * 128;

  v8f acc[4][2];
#pragma unroll
  for (int i = 0; i < 4; ++i)
#pragma unroll
    for (int j = 0; j < 2; ++j) acc[i][j] = zero8();

  for (int k0 = 0; k0 < Kdim; k0 += 64) {
    // ---- stage A tile (128 x 64) ----
    if (ABF16) {
      const __bf16* A16 = (const __bf16*)Ap;
#pragma unroll
      for (int p = 0; p < 4; ++p) {
        int idx = tid + p * 256;
        int r = idx >> 3, cg = (idx & 7) * 8;
        const __bf16* src = A16 + (size_t)(Mbase + r) * Kdim + k0 + cg;
#if HAVE_ASYNC_LDS
        async_cp_b128(src, &As[r * LD + cg]);     // DMA, bypasses VGPRs
#else
        *(uint4*)(&As[r * LD + cg]) = *(const uint4*)src;
#endif
      }
    } else {
      const float* A32 = (const float*)Ap;
#pragma unroll
      for (int p = 0; p < 8; ++p) {
        int idx = tid + p * 256;
        int r = idx >> 4, cg = (idx & 15) * 4;
        float4 f = *(const float4*)(A32 + (size_t)(Mbase + r) * Kdim + k0 + cg);
        __bf16* d = &As[r * LD + cg];
        d[0] = f2bf(f.x); d[1] = f2bf(f.y); d[2] = f2bf(f.z); d[3] = f2bf(f.w);
      }
    }
    // ---- stage B tile (64 x 128), transposed to [n][k] ----
#pragma unroll
    for (int p = 0; p < 8; ++p) {
      int idx = tid + p * 256;
      int r = idx >> 5, cg = (idx & 31) * 4;
      float4 f = *(const float4*)(W + (size_t)(k0 + r) * Ndim + Nbase + cg);
      Bs[(cg + 0) * LD + r] = f2bf(f.x);
      Bs[(cg + 1) * LD + r] = f2bf(f.y);
      Bs[(cg + 2) * LD + r] = f2bf(f.z);
      Bs[(cg + 3) * LD + r] = f2bf(f.w);
    }
    if (ABF16) wait_async0();
    __syncthreads();

    // prefetch next K tiles into L2 while the math runs
    if (k0 + 64 < Kdim) {
      if (!ABF16)
        __builtin_prefetch((const float*)Ap +
                           (size_t)(Mbase + (tid >> 1)) * Kdim + k0 + 64, 0, 1);
      __builtin_prefetch(W + (size_t)(k0 + 64 + (tid >> 2)) * Ndim + Nbase, 0, 1);
    }

#pragma unroll
    for (int ks = 0; ks < 2; ++ks) {
      const int kb = ks * 32;
      v16bf af[4], bfr[2];
#pragma unroll
      for (int i = 0; i < 4; ++i)
        af[i] = load_frag_a(&As[(wrow * 64 + i * 16 + l15) * LD + kb], hi);
#pragma unroll
      for (int j = 0; j < 2; ++j)
        bfr[j] = load_frag_b(&Bs[(wcol * 32 + j * 16 + l15) * LD + kb], hi);
#pragma unroll
      for (int i = 0; i < 4; ++i)
#pragma unroll
        for (int j = 0; j < 2; ++j)
          acc[i][j] = wmma_bf16(af[i], bfr[j], acc[i][j]);
    }
    __syncthreads();
  }

  // ---- epilogue ----
#pragma unroll
  for (int i = 0; i < 4; ++i) {
#pragma unroll
    for (int j = 0; j < 2; ++j) {
      const int n = Nbase + wcol * 32 + j * 16 + l15;
      const float bv = bias[n];
#pragma unroll
      for (int r = 0; r < 8; ++r) {
        const int m = Mbase + wrow * 64 + i * 16 + r + (hi ? 8 : 0);
        const float val = acc[i][j][r] + bv;
        if (HEADSPLIT) {   // -> bf16 [b][h][t][dk]
          const int b = m >> 11, t = m & 2047;
          const int h = n >> 6, dk = n & 63;
          ((__bf16*)outp)[(((size_t)(b * 16 + h)) * 2048 + t) * 64 + dk] =
              f2bf(val);
        } else {           // -> fp32 [m][n]
          ((float*)outp)[(size_t)m * Ndim + n] = val;
        }
      }
    }
  }
}

// ---------------------------------------------------------------------------
// Flash attention: one block = 128 query rows of one (b,h); 8 waves x 16 rows.
// S processed in chunks of 64 with causal early exit. Scores and PV both via
// WMMA bf16; online softmax state per lane (8 rows each, split by lane half).
// K tiles staged with async load-to-LDS when available; V staged through
// VGPRs because it must be transposed to [dk][s] for the PV B-fragments.
// ---------------------------------------------------------------------------
__global__ __launch_bounds__(256)
void attn_wmma(const void* __restrict__ Qp, const void* __restrict__ Kp,
               const void* __restrict__ Vp, const int* __restrict__ mask,
               const float* __restrict__ sbias, const float* __restrict__ sbs_p,
               void* __restrict__ AOp) {
  constexpr int LD = 72;
  __shared__ __bf16 Ks[64 * LD];        // [s][dk]   (B-frag layout for QK^T)
  __shared__ __bf16 Vt[64 * LD];        // [dk][s]   (B-frag layout for PV)
  __shared__ __bf16 Pb[8 * 16 * LD];    // per-wave P slab, [row][s]

  const int tid  = threadIdx.x;
  const int lane = tid & 31, wave = tid >> 5;
  const int l15  = lane & 15, hi = lane >> 4;
  const int b    = blockIdx.y >> 4, h = blockIdx.y & 15;
  const int q0   = blockIdx.x * 128;
  const float sbs = *sbs_p;

  const __bf16* Qh = (const __bf16*)Qp + ((size_t)(b * 16 + h)) * 2048 * 64;
  const __bf16* Kh = (const __bf16*)Kp + ((size_t)(b * 16 + h)) * 2048 * 64;
  const __bf16* Vh = (const __bf16*)Vp + ((size_t)(b * 16 + h)) * 2048 * 64;

  // Q A-fragments straight from global (rows are contiguous 128B)
  const int qrow = q0 + wave * 16 + l15;
  v16bf qf[2];
#pragma unroll
  for (int kc = 0; kc < 2; ++kc)
    qf[kc] = load_frag_a(Qh + (size_t)qrow * 64 + kc * 32, hi);

  float m8[8], l8[8];
  v8f oacc[4];
#pragma unroll
  for (int r = 0; r < 8; ++r) { m8[r] = -3.0e38f; l8[r] = 0.f; }
#pragma unroll
  for (int j = 0; j < 4; ++j) oacc[j] = zero8();

  const int nchunks = 2 * blockIdx.x + 2;   // causal: only s <= q_max chunks
  for (int c = 0; c < nchunks; ++c) {
    const int s0 = c * 64;
    // ---- stage K (async DMA when available) and V (VGPR transpose) ----
#pragma unroll
    for (int p = 0; p < 2; ++p) {
      int idx = tid + p * 256;
      int r = idx >> 3, cg = (idx & 7) * 8;
      const __bf16* ksrc = Kh + (size_t)(s0 + r) * 64 + cg;
#if HAVE_ASYNC_LDS
      async_cp_b128(ksrc, &Ks[r * LD + cg]);
#else
      *(uint4*)(&Ks[r * LD + cg]) = *(const uint4*)ksrc;
#endif
      union { uint4 u; __bf16 e[8]; } vu;
      vu.u = *(const uint4*)(Vh + (size_t)(s0 + r) * 64 + cg);
#pragma unroll
      for (int i = 0; i < 8; ++i) Vt[(cg + i) * LD + r] = vu.e[i];
    }
    wait_async0();
    __syncthreads();

    // ---- scores: 4 tiles of 16x16, dk accumulated in two K=32 steps ----
    v8f sc[4];
#pragma unroll
    for (int j = 0; j < 4; ++j) {
      sc[j] = zero8();
#pragma unroll
      for (int kc = 0; kc < 2; ++kc) {
        v16bf kb = load_frag_b(&Ks[(j * 16 + l15) * LD + kc * 32], hi);
        sc[j] = wmma_bf16(qf[kc], kb, sc[j]);
      }
    }

    // ---- scale + sentence bias + padding mask + causal ----
#pragma unroll
    for (int j = 0; j < 4; ++j) {
      const int scol = s0 + j * 16 + l15;
      const int mk = mask[b * 2048 + scol];
      const float sb = sbs * sbias[b * 2048 + scol];
#pragma unroll
      for (int r = 0; r < 8; ++r) {
        const int qr = q0 + wave * 16 + r + (hi ? 8 : 0);
        float v = sc[j][r] * 0.125f + sb;               // 1/sqrt(64)
        if (mk == 0)   v = -1e9f;
        if (scol > qr) v = -1e9f;
        sc[j][r] = v;
      }
    }

    // ---- online softmax update (rows live across 16-lane halves) ----
#pragma unroll
    for (int r = 0; r < 8; ++r) {
      float cm = fmaxf(fmaxf(sc[0][r], sc[1][r]), fmaxf(sc[2][r], sc[3][r]));
      cm = fmaxf(cm, __shfl_xor(cm, 1, 32));
      cm = fmaxf(cm, __shfl_xor(cm, 2, 32));
      cm = fmaxf(cm, __shfl_xor(cm, 4, 32));
      cm = fmaxf(cm, __shfl_xor(cm, 8, 32));
      const float mnew  = fmaxf(m8[r], cm);
      const float alpha = __expf(m8[r] - mnew);
      float ps = 0.f;
#pragma unroll
      for (int j = 0; j < 4; ++j) {
        const float pv = __expf(sc[j][r] - mnew);
        sc[j][r] = pv;
        ps += pv;
      }
      ps += __shfl_xor(ps, 1, 32);
      ps += __shfl_xor(ps, 2, 32);
      ps += __shfl_xor(ps, 4, 32);
      ps += __shfl_xor(ps, 8, 32);
      l8[r] = l8[r] * alpha + ps;
      m8[r] = mnew;
#pragma unroll
      for (int j = 0; j < 4; ++j) oacc[j][r] *= alpha;
    }

    // ---- C-layout -> A-layout via wave-private LDS slab ----
    __bf16* Pw = &Pb[wave * 16 * LD];
#pragma unroll
    for (int j = 0; j < 4; ++j)
#pragma unroll
      for (int r = 0; r < 8; ++r)
        Pw[(r + (hi ? 8 : 0)) * LD + j * 16 + l15] = f2bf(sc[j][r]);
    asm volatile("s_wait_dscnt 0" ::: "memory");  // wave-local LDS RAW fence

    // ---- PV: out[16 x 64] += P[16 x 64] x V[64 x 64] ----
#pragma unroll
    for (int kc = 0; kc < 2; ++kc) {
      v16bf pa = load_frag_a(&Pw[l15 * LD + kc * 32], hi);
#pragma unroll
      for (int j = 0; j < 4; ++j) {
        v16bf vb = load_frag_b(&Vt[(j * 16 + l15) * LD + kc * 32], hi);
        oacc[j] = wmma_bf16(pa, vb, oacc[j]);
      }
    }
    __syncthreads();   // before next chunk overwrites Ks/Vt
  }

  // ---- normalize and write AO (bf16, heads concatenated along D) ----
  __bf16* AO = (__bf16*)AOp;
#pragma unroll
  for (int r = 0; r < 8; ++r) {
    const int t = q0 + wave * 16 + r + (hi ? 8 : 0);
    const float inv = 1.f / fmaxf(l8[r], 1e-30f);
#pragma unroll
    for (int j = 0; j < 4; ++j) {
      const int col = h * 64 + j * 16 + l15;
      AO[((size_t)b * 2048 + t) * 1024 + col] = f2bf(oacc[j][r] * inv);
    }
  }
}

// ---------------------------------------------------------------------------
extern "C" void kernel_launch(void* const* d_in, const int* in_sizes, int n_in,
                              void* d_out, int out_size, void* d_ws,
                              size_t ws_size, hipStream_t stream) {
  (void)in_sizes; (void)n_in; (void)out_size; (void)ws_size;
  const float* q     = (const float*)d_in[0];
  const float* k     = (const float*)d_in[1];
  const float* v     = (const float*)d_in[2];
  const int*   mask  = (const int*)d_in[3];
  // d_in[4] = causal_mask (tril) — computed analytically in-kernel.
  const float* sbias = (const float*)d_in[5];
  const float* Wq = (const float*)d_in[6];  const float* bq = (const float*)d_in[7];
  const float* Wk = (const float*)d_in[8];  const float* bk = (const float*)d_in[9];
  const float* Wv = (const float*)d_in[10]; const float* bv = (const float*)d_in[11];
  const float* Wo = (const float*)d_in[12]; const float* bo = (const float*)d_in[13];
  const float* sbs = (const float*)d_in[14];

  char* ws = (char*)d_ws;
  void* Qb = (void*)(ws);
  void* Kb = (void*)(ws + (size_t)8  * 1024 * 1024);
  void* Vb = (void*)(ws + (size_t)16 * 1024 * 1024);
  void* AO = (void*)(ws + (size_t)24 * 1024 * 1024);

  const dim3 gg(8, 32), gb(256);                 // GEMM: N/128 x M/128
  gemm_wmma<0, 1><<<gg, gb, 0, stream>>>(q, Wq, bq, Qb);
  gemm_wmma<0, 1><<<gg, gb, 0, stream>>>(k, Wk, bk, Kb);
  gemm_wmma<0, 1><<<gg, gb, 0, stream>>>(v, Wv, bv, Vb);
  attn_wmma<<<dim3(16, 32), 256, 0, stream>>>(Qb, Kb, Vb, mask, sbias, sbs, AO);
  gemm_wmma<1, 0><<<gg, gb, 0, stream>>>(AO, Wo, bo, d_out);
}
